// SPVCNN_SWIFTNET_ASSOP2P_68152541053517
// MI455X (gfx1250) — matrix operand order repaired and checked by reading, not compiled, not run on hardware
//
#include <hip/hip_runtime.h>
#include <hip/hip_bf16.h>
#include <hip/hip_fp16.h>

#define MVOX 200000
#define BB 2
#define VV 6
#define NPP 120000
#define NPTS (BB*NPP)        // 240000
#define CIN 128
#define CPF 64
#define RCH 16
#define HHH 160
#define WWW 320
#define HWPIX (HHH*WWW)      // 51200

typedef _Float16 h8   __attribute__((ext_vector_type(8)));
typedef _Float16 v16h __attribute__((ext_vector_type(16)));
typedef float    v8f  __attribute__((ext_vector_type(8)));

__device__ __forceinline__ v8f zero8() {
  v8f c;
#pragma unroll
  for (int i = 0; i < 8; ++i) c[i] = 0.0f;
  return c;
}

// ---------------- workspace zeroing (sums + counts) ----------------
__global__ void zero_f4(float4* __restrict__ p, int n4) {
  for (int i = blockIdx.x * blockDim.x + threadIdx.x; i < n4;
       i += gridDim.x * blockDim.x)
    p[i] = make_float4(0.f, 0.f, 0.f, 0.f);
}

// ---------------- point -> voxel scatter (atomic segment sum) ----------------
__global__ void scatter_p2v(const float* __restrict__ pf, const int* __restrict__ idx,
                            float* __restrict__ sums, float* __restrict__ cnt) {
  int t = blockIdx.x * 256 + threadIdx.x;
  int p = t >> 6, c = t & 63;
  int v = idx[p];
  __hip_atomic_fetch_add(&sums[(size_t)v * CPF + c], pf[t],
                         __ATOMIC_RELAXED, __HIP_MEMORY_SCOPE_AGENT);
  if (c == 0)
    __hip_atomic_fetch_add(&cnt[v], 1.0f,
                           __ATOMIC_RELAXED, __HIP_MEMORY_SCOPE_AGENT);
}

// ------------- img NCHW f32 -> NHWC f16 (157MB -> fits in 192MB L2) -------------
__global__ void transpose_img(const float* __restrict__ img, _Float16* __restrict__ imgT) {
  __shared__ float tile[32][33];
  int tx = threadIdx.x & 31, ty = threadIdx.x >> 5;
  int pix0 = blockIdx.x * 32, ci0 = blockIdx.y * 32, bv = blockIdx.z;
  const float* src = img + ((size_t)bv * CIN + ci0) * HWPIX + pix0;
#pragma unroll
  for (int r = 0; r < 4; ++r) {
    int c = ty + r * 8;
    tile[c][tx] = src[(size_t)c * HWPIX + tx];   // coalesced along pixels
  }
  __syncthreads();
  _Float16* dst = imgT + ((size_t)bv * HWPIX + pix0) * CIN + ci0;
#pragma unroll
  for (int r = 0; r < 4; ++r) {
    int p = ty + r * 8;
    dst[(size_t)p * CIN + tx] = (_Float16)tile[tx][p];  // coalesced along channels
  }
}

// ---- one-shot: transpose tiny weight matrices to f16 column-major [N][K] ----
__global__ void prep_weights(const float* __restrict__ w_fc1, const float* __restrict__ w_fc2,
                             const float* __restrict__ w_conv1, const float* __restrict__ w_fuse,
                             _Float16* __restrict__ fc1T, _Float16* __restrict__ fc2T,
                             _Float16* __restrict__ conv1T, _Float16* __restrict__ fuseT) {
  int t = blockIdx.x * 256 + threadIdx.x;
  int stride = gridDim.x * 256;
  for (int i = t; i < CIN * RCH; i += stride) {          // w_fc1 [128][16]
    int k = i >> 4, n = i & 15;
    fc1T[n * CIN + k] = (_Float16)w_fc1[i];
  }
  for (int i = t; i < CPF * RCH; i += stride) {          // w_fc2 [64][16]
    int k = i >> 4, n = i & 15;
    fc2T[n * CPF + k] = (_Float16)w_fc2[i];
  }
  for (int i = t; i < CIN * CPF; i += stride) {          // w_conv1 [128][64]
    int k = i >> 6, n = i & 63;
    conv1T[n * CIN + k] = (_Float16)w_conv1[i];
  }
  for (int i = t; i < CIN * CPF; i += stride) {          // w_fuse [128][64]
    int k = i >> 6, n = i & 63;
    fuseT[n * CIN + k] = (_Float16)w_fuse[i];
  }
}

// ------- voxel -> point trilinear gather, emits f16 rows (8 ch / thread) -------
__global__ void v2p_gather(const float* __restrict__ sums, const float* __restrict__ cnt,
                           const int* __restrict__ vidx, const float* __restrict__ vw,
                           _Float16* __restrict__ pts) {
  int t = blockIdx.x * 256 + threadIdx.x;
  int p = t >> 3, c0 = (t & 7) * 8;
  float acc[8];
#pragma unroll
  for (int i = 0; i < 8; ++i) acc[i] = 0.f;
#pragma unroll
  for (int j = 0; j < 8; ++j) {
    int vi = vidx[p * 8 + j];
    float s = vw[p * 8 + j] / fmaxf(cnt[vi], 1.0f);
    const float4* sp = (const float4*)(sums + (size_t)vi * CPF + c0);
    float4 a = sp[0], b = sp[1];
    acc[0] += s * a.x; acc[1] += s * a.y; acc[2] += s * a.z; acc[3] += s * a.w;
    acc[4] += s * b.x; acc[5] += s * b.y; acc[6] += s * b.z; acc[7] += s * b.w;
  }
  h8 o;
#pragma unroll
  for (int i = 0; i < 8; ++i) o[i] = (_Float16)acc[i];
  *(h8*)&pts[(size_t)p * CPF + c0] = o;
}

// ---------------- WMMA fragment helpers ----------------
// A or B frag from a row-major f16 row (LDS or global): two contiguous 16B loads.
// A: lane holds row M=lane&15; e 0..7 -> K = base+hi*8+e, e 8..15 -> K = base+16+hi*8+(e-8)
__device__ __forceinline__ v16h a_frag_h16(const _Float16* rowp, int kc, int hi) {
  h8 a0 = *(const h8*)(rowp + kc * 32 + hi * 8);
  h8 a1 = *(const h8*)(rowp + kc * 32 + 16 + hi * 8);
  return __builtin_shufflevector(a0, a1, 0, 1, 2, 3, 4, 5, 6, 7,
                                 8, 9, 10, 11, 12, 13, 14, 15);
}
// B frag from f16 column-major wT[N][K]: lane = column n, e 0..7 -> k = kbase + 8*hi + e, e 8..15 -> k = kbase + 16 + 8*hi + (e-8)
__device__ __forceinline__ v16h b_frag16(const _Float16* __restrict__ wT, int Kdim,
                                         int kbase, int col, int hi) {
  const _Float16* p = wT + (size_t)col * Kdim + kbase + 8 * hi;
  h8 b0 = *(const h8*)p;
  h8 b1 = *(const h8*)(p + 16);
  return __builtin_shufflevector(b0, b1, 0, 1, 2, 3, 4, 5, 6, 7,
                                 8, 9, 10, 11, 12, 13, 14, 15);
}

// ---------- fused: bilinear-gather + IA-Layer attention + GEMMs (WMMA) ----------
__global__ void __launch_bounds__(256) fused_attn(
    const _Float16* __restrict__ imgT, const float* __restrict__ coords,
    const unsigned char* __restrict__ masks, const _Float16* __restrict__ pts,
    const _Float16* __restrict__ fc1T, const float* __restrict__ b_fc1,
    const _Float16* __restrict__ fc2T, const float* __restrict__ b_fc2,
    const float* __restrict__ w_fc3, const float* __restrict__ b_fc3,
    const _Float16* __restrict__ conv1T, const float* __restrict__ b_conv1,
    const _Float16* __restrict__ fuseT, const float* __restrict__ b_fuse,
    float* __restrict__ out) {
  __shared__ __align__(16) _Float16 sBuf[8][16 * 128];  // per-wave 16x128 f16 tile
  const int wv = threadIdx.x >> 5;
  const int lane = threadIdx.x & 31;
  const int row = lane & 15;   // A row / B-C-D column index
  const int hi = lane >> 4;    // half-wave selector
  const int tile = blockIdx.x * 8 + wv;
  _Float16* sb = sBuf[wv];

  const int p = tile * 16 + row;     // this lane's point
  const int b = p / NPP;
  const int pp = p - b * NPP;

  // ---- phase 1: build imf tile in LDS via bilinear sampling (last masked view) ----
  int vstar = -1;
#pragma unroll
  for (int v = VV - 1; v >= 0; --v)
    if (vstar < 0 && masks[((size_t)b * VV + v) * NPP + pp]) vstar = v;
  const int ci0 = hi * 64;
  if (vstar >= 0) {
    const size_t cb = (((size_t)b * VV + vstar) * NPP + pp) * 2;
    float cx = coords[cb], cy = coords[cb + 1];
    float x = (cx + 1.f) * 0.5f * (WWW - 1);
    float y = (cy + 1.f) * 0.5f * (HHH - 1);
    float fx = floorf(x), fy = floorf(y);
    float wx1 = x - fx, wx0 = 1.f - wx1;
    float wy1 = y - fy, wy0 = 1.f - wy1;
    int x0 = min(max((int)fx, 0), WWW - 1); int x1 = min(x0 + 1, WWW - 1);
    int y0 = min(max((int)fy, 0), HHH - 1); int y1 = min(y0 + 1, HHH - 1);
    const _Float16* base = imgT + (size_t)(b * VV + vstar) * HWPIX * CIN;
    const _Float16* p00 = base + ((size_t)y0 * WWW + x0) * CIN + ci0;
    const _Float16* p01 = base + ((size_t)y0 * WWW + x1) * CIN + ci0;
    const _Float16* p10 = base + ((size_t)y1 * WWW + x0) * CIN + ci0;
    const _Float16* p11 = base + ((size_t)y1 * WWW + x1) * CIN + ci0;
    // packed-f16 bilinear: lowers to v_pk_mul_f16 / v_pk_fma_f16 (co-executes with WMMA)
    const _Float16 hw00 = (_Float16)(wy0 * wx0), hw01 = (_Float16)(wy0 * wx1);
    const _Float16 hw10 = (_Float16)(wy1 * wx0), hw11 = (_Float16)(wy1 * wx1);
#pragma unroll
    for (int c = 0; c < 64; c += 8) {
      h8 t00 = *(const h8*)(p00 + c); h8 t01 = *(const h8*)(p01 + c);
      h8 t10 = *(const h8*)(p10 + c); h8 t11 = *(const h8*)(p11 + c);
      h8 o = t00 * hw00 + t01 * hw01 + t10 * hw10 + t11 * hw11;
      *(h8*)&sb[row * 128 + ci0 + c] = o;
    }
  } else {
    h8 z;
#pragma unroll
    for (int i = 0; i < 8; ++i) z[i] = (_Float16)0.f;
#pragma unroll
    for (int c = 0; c < 64; c += 8) *(h8*)&sb[row * 128 + ci0 + c] = z;
  }
  __syncthreads();

  // ---- A fragments ----
  v16h Aimf[4];
#pragma unroll
  for (int kc = 0; kc < 4; ++kc) Aimf[kc] = a_frag_h16(sb + row * 128, kc, hi);
  const _Float16* ptsRow = pts + (size_t)p * CPF;
  v16h Apts[2];
#pragma unroll
  for (int kc = 0; kc < 2; ++kc) Apts[kc] = a_frag_h16(ptsRow, kc, hi);
  __syncthreads();  // all lanes done reading imf tile before it is overwritten

  // ---- conv1: imf @ w_conv1 -> 16x64 (4 N-tiles) ----
  v8f accC[4];
#pragma unroll
  for (int nt = 0; nt < 4; ++nt) {
    v8f c = zero8();
#pragma unroll
    for (int kc = 0; kc < 4; ++kc) {
      v16h bf = b_frag16(conv1T, CIN, kc * 32, nt * 16 + row, hi);
      c = __builtin_amdgcn_wmma_f32_16x16x32_f16(false, Aimf[kc], false, bf,
                                                 (short)0, c, false, false);
    }
    accC[nt] = c;
  }
  // ---- ri + rp = imf@w_fc1 + pts@w_fc2 ----
  v8f accR = zero8();
#pragma unroll
  for (int kc = 0; kc < 4; ++kc) {
    v16h bf = b_frag16(fc1T, CIN, kc * 32, row, hi);
    accR = __builtin_amdgcn_wmma_f32_16x16x32_f16(false, Aimf[kc], false, bf,
                                                  (short)0, accR, false, false);
  }
#pragma unroll
  for (int kc = 0; kc < 2; ++kc) {
    v16h bf = b_frag16(fc2T, CPF, kc * 32, row, hi);
    accR = __builtin_amdgcn_wmma_f32_16x16x32_f16(false, Apts[kc], false, bf,
                                                  (short)0, accR, false, false);
  }
  // ---- attention: sigmoid(tanh(ri+rp) @ w_fc3 + b3), reduce over RC via shfl ----
  const float bb12 = b_fc1[row] + b_fc2[row];
  const float w3 = w_fc3[row];
  const float b3 = b_fc3[0];
  float att[8];
#pragma unroll
  for (int r = 0; r < 8; ++r) {
    float t = tanhf(accR[r] + bb12);
    float pr = t * w3;
    pr += __shfl_xor(pr, 1, 16);
    pr += __shfl_xor(pr, 2, 16);
    pr += __shfl_xor(pr, 4, 16);
    pr += __shfl_xor(pr, 8, 16);
    att[r] = 1.f / (1.f + expf(-(pr + b3)));
  }
  // ---- build concat(pts, img_t) tile in LDS as f16 ----
#pragma unroll
  for (int c = 0; c < 32; c += 8)
    *(h8*)&sb[row * 128 + hi * 32 + c] = *(const h8*)(ptsRow + hi * 32 + c);
#pragma unroll
  for (int nt = 0; nt < 4; ++nt) {
    float bc = b_conv1[nt * 16 + row];
#pragma unroll
    for (int r = 0; r < 8; ++r) {
      float v = (accC[nt][r] + bc) * att[r];
      sb[(r + 8 * hi) * 128 + 64 + nt * 16 + row] = (_Float16)v;
    }
  }
  __syncthreads();
  // ---- fused = relu(concat @ w_fuse + b_fuse) ----
  v16h Af[4];
#pragma unroll
  for (int kc = 0; kc < 4; ++kc) Af[kc] = a_frag_h16(sb + row * 128, kc, hi);
#pragma unroll
  for (int nt = 0; nt < 4; ++nt) {
    v8f c = zero8();
#pragma unroll
    for (int kc = 0; kc < 4; ++kc) {
      v16h bf = b_frag16(fuseT, CIN, kc * 32, nt * 16 + row, hi);
      c = __builtin_amdgcn_wmma_f32_16x16x32_f16(false, Af[kc], false, bf,
                                                 (short)0, c, false, false);
    }
    float bfu = b_fuse[nt * 16 + row];
#pragma unroll
    for (int r = 0; r < 8; ++r) {
      float v = fmaxf(c[r] + bfu, 0.f);
      out[(size_t)(tile * 16 + r + 8 * hi) * CPF + nt * 16 + row] = v;
    }
  }
}

extern "C" void kernel_launch(void* const* d_in, const int* in_sizes, int n_in,
                              void* d_out, int out_size, void* d_ws, size_t ws_size,
                              hipStream_t stream) {
  (void)in_sizes; (void)n_in; (void)out_size; (void)ws_size;
  const float* points_feat = (const float*)d_in[0];
  const float* img     = (const float*)d_in[1];
  const float* coords  = (const float*)d_in[2];
  const float* v2p_w   = (const float*)d_in[3];
  const float* w_fc1   = (const float*)d_in[4];
  const float* b_fc1   = (const float*)d_in[5];
  const float* w_fc2   = (const float*)d_in[6];
  const float* b_fc2   = (const float*)d_in[7];
  const float* w_fc3   = (const float*)d_in[8];
  const float* b_fc3   = (const float*)d_in[9];
  const float* w_conv1 = (const float*)d_in[10];
  const float* b_conv1 = (const float*)d_in[11];
  const float* w_fuse  = (const float*)d_in[12];
  const float* b_fuse  = (const float*)d_in[13];
  const unsigned char* masks = (const unsigned char*)d_in[14];
  const int* pt2vox  = (const int*)d_in[15];
  const int* v2p_idx = (const int*)d_in[16];
  float* out = (float*)d_out;

  char* ws = (char*)d_ws;
  float* sums    = (float*)ws;                              // M*64 f32 = 51.2 MB
  float* cnt     = (float*)(ws + (size_t)MVOX * CPF * 4);   // M f32
  _Float16* pts  = (_Float16*)(ws + 52000000);              // 240000*64 f16 = 30.7 MB
  _Float16* imgT = (_Float16*)(ws + 113440000);             // 12*160*320*128 f16 = 157 MB
  char* wbase    = ws + 270726400;
  _Float16* conv1T = (_Float16*)(wbase);                    // 64*128 f16
  _Float16* fuseT  = (_Float16*)(wbase + 16384);            // 64*128 f16
  _Float16* fc1T   = (_Float16*)(wbase + 32768);            // 16*128 f16
  _Float16* fc2T   = (_Float16*)(wbase + 36864);            // 16*64  f16

  int n4 = (MVOX * CPF + MVOX) / 4;
  zero_f4<<<4096, 256, 0, stream>>>((float4*)sums, n4);
  scatter_p2v<<<(NPTS * CPF) / 256, 256, 0, stream>>>(points_feat, pt2vox, sums, cnt);
  dim3 tg(HWPIX / 32, CIN / 32, BB * VV);
  transpose_img<<<tg, 256, 0, stream>>>(img, imgT);
  prep_weights<<<32, 256, 0, stream>>>(w_fc1, w_fc2, w_conv1, w_fuse,
                                       fc1T, fc2T, conv1T, fuseT);
  v2p_gather<<<(NPTS * 8) / 256, 256, 0, stream>>>(sums, cnt, v2p_idx, v2p_w, pts);
  fused_attn<<<(NPTS / 16) / 8, 256, 0, stream>>>(
      imgT, coords, masks, pts, fc1T, b_fc1, fc2T, b_fc2, w_fc3, b_fc3,
      conv1T, b_conv1, fuseT, b_fuse, out);
}
